// model_74741020885579
// MI455X (gfx1250) — matrix operand order repaired and checked by reading, not compiled, not run on hardware
//
#include <hip/hip_runtime.h>
#include <hip/hip_bf16.h>
#include <math.h>

// ---------------- problem constants ----------------
constexpr int H  = 128;
constexpr int T  = 1024;
constexpr int V  = 50257;
constexpr int G4 = 4 * H;                 // 512 gate rows
constexpr int NT = (V + 15) / 16;         // 3142 N-tiles
constexpr int VP = NT * 16;               // 50272 padded vocab
constexpr int MT = T / 16;                // 64 M-tiles

typedef _Float16 v16h __attribute__((ext_vector_type(16)));
typedef _Float16 v8h  __attribute__((ext_vector_type(8)));
typedef float    v8f  __attribute__((ext_vector_type(8)));

__device__ __forceinline__ float sigf(float x) { return 1.0f / (1.0f + expf(-x)); }

// ---------------- encoder scan: 1 persistent workgroup ----------------
__global__ __launch_bounds__(512) void enc_kernel(
    const int* __restrict__ seq, const float* __restrict__ embeds,
    const float* __restrict__ Wih, const float* __restrict__ Whh,
    const float* __restrict__ b,
    float* __restrict__ enc_out, float* __restrict__ h0c0)
{
  __shared__ alignas(16) float x[H], h[H], c[H];
  __shared__ float gates[G4];
  const int tid = threadIdx.x;
  if (tid < H) { h[tid] = 0.0f; c[tid] = 0.0f; }
  __syncthreads();

  for (int t = 0; t < T; ++t) {
    if (tid < H) x[tid] = embeds[(size_t)seq[t] * H + tid];
    __syncthreads();
    {
      const float4* wi = (const float4*)(Wih + (size_t)tid * H);
      const float4* wh = (const float4*)(Whh + (size_t)tid * H);
      const float4* xv = (const float4*)x;
      const float4* hv = (const float4*)h;
      float acc = b[tid];
      #pragma unroll 8
      for (int q = 0; q < H / 4; ++q) {
        float4 a = wi[q], xx = xv[q];
        acc += a.x * xx.x + a.y * xx.y + a.z * xx.z + a.w * xx.w;
      }
      #pragma unroll 8
      for (int q = 0; q < H / 4; ++q) {
        float4 a = wh[q], hh = hv[q];
        acc += a.x * hh.x + a.y * hh.y + a.z * hh.z + a.w * hh.w;
      }
      gates[tid] = acc;
    }
    __syncthreads();
    if (tid < H) {
      float ig = sigf(gates[tid]);
      float fg = sigf(gates[H + tid]);
      float gg = tanhf(gates[2 * H + tid]);
      float og = sigf(gates[3 * H + tid]);
      float cn = fg * c[tid] + ig * gg;
      float hn = og * tanhf(cn);
      c[tid] = cn; h[tid] = hn;
      enc_out[(size_t)t * H + tid] = hn;
    }
    __syncthreads();
  }
  if (tid < H) { h0c0[tid] = h[tid]; h0c0[H + tid] = c[tid]; }
}

// ---------------- encW1 = enc_out @ W1  (tiny GEMM) ----------------
__global__ __launch_bounds__(H) void encw1_kernel(
    const float* __restrict__ enc_out, const float* __restrict__ W1,
    float* __restrict__ encW1)
{
  const int t = blockIdx.x, k = threadIdx.x;
  __shared__ alignas(16) float e[H];
  e[k] = enc_out[(size_t)t * H + k];
  __syncthreads();
  float acc = 0.0f;
  #pragma unroll 8
  for (int i = 0; i < H; ++i) acc += e[i] * W1[(size_t)i * H + k];  // coalesced in k
  encW1[(size_t)t * H + k] = acc;
}

// ---------------- decoder scan: 1 persistent workgroup ----------------
__global__ __launch_bounds__(512) void dec_kernel(
    const int* __restrict__ gold, const float* __restrict__ embeds,
    const float* __restrict__ Wih, const float* __restrict__ Whh,
    const float* __restrict__ b,   const float* __restrict__ W2,
    const float* __restrict__ v,   const float* __restrict__ enc_out,
    const float* __restrict__ encW1, const float* __restrict__ h0c0,
    _Float16* __restrict__ h16)
{
  __shared__ alignas(16) float h[H], c[H], hW2[H], vsh[H], xbuf[2 * H];
  __shared__ float gates[G4];
  __shared__ float scores[T];
  __shared__ float red[512];
  const int tid = threadIdx.x;

  if (tid < H) { h[tid] = h0c0[tid]; c[tid] = h0c0[H + tid]; vsh[tid] = v[tid]; }
  __syncthreads();

  for (int t = 0; t < T; ++t) {
    // hW2[k] = sum_i h[i] * W2[i][k]
    if (tid < H) {
      float acc = 0.0f;
      #pragma unroll 8
      for (int i = 0; i < H; ++i) acc += h[i] * W2[(size_t)i * H + tid];
      hW2[tid] = acc;
    }
    __syncthreads();

    // scores[r] = tanh(encW1[r] + hW2) . v   (2 rows / thread)
    for (int r = tid; r < T; r += 512) {
      const float* e = encW1 + (size_t)r * H;
      float s = 0.0f;
      #pragma unroll 8
      for (int k = 0; k < H; ++k) s += tanhf(e[k] + hW2[k]) * vsh[k];
      scores[r] = s;
    }
    __syncthreads();

    // softmax over T=1024
    float s0 = scores[tid], s1 = scores[tid + 512];
    red[tid] = fmaxf(s0, s1);
    __syncthreads();
    for (int sft = 256; sft > 0; sft >>= 1) {
      if (tid < sft) red[tid] = fmaxf(red[tid], red[tid + sft]);
      __syncthreads();
    }
    const float mx = red[0];
    __syncthreads();
    float e0 = expf(s0 - mx), e1 = expf(s1 - mx);
    scores[tid] = e0; scores[tid + 512] = e1;
    red[tid] = e0 + e1;
    __syncthreads();
    for (int sft = 256; sft > 0; sft >>= 1) {
      if (tid < sft) red[tid] += red[tid + sft];
      __syncthreads();
    }
    const float inv = 1.0f / red[0];
    __syncthreads();

    // context[k] = sum_r attn[r] * enc_out[r][k]  (4 partial groups)
    {
      const int k = tid & (H - 1);
      const int grp = tid >> 7;           // 0..3, each covers 256 rows
      float acc = 0.0f;
      const int r0 = grp * 256;
      for (int r = r0; r < r0 + 256; ++r) acc += scores[r] * enc_out[(size_t)r * H + k];
      red[tid] = acc;
      __syncthreads();
      if (tid < H) {
        float ctx = (red[tid] + red[tid + 128] + red[tid + 256] + red[tid + 384]) * inv;
        xbuf[H + tid] = fmaxf(ctx, 0.0f);
        float pe = (t == 0) ? 0.0f : embeds[(size_t)gold[t - 1] * H + tid];
        xbuf[tid] = fmaxf(pe, 0.0f);
      }
    }
    __syncthreads();

    // decoder LSTM gates: Wih_dec (512 x 256), Whh_dec (512 x 128)
    {
      const float4* wi = (const float4*)(Wih + (size_t)tid * 2 * H);
      const float4* wh = (const float4*)(Whh + (size_t)tid * H);
      const float4* xv = (const float4*)xbuf;
      const float4* hv = (const float4*)h;
      float acc = b[tid];
      #pragma unroll 8
      for (int q = 0; q < 2 * H / 4; ++q) {
        float4 a = wi[q], xx = xv[q];
        acc += a.x * xx.x + a.y * xx.y + a.z * xx.z + a.w * xx.w;
      }
      #pragma unroll 8
      for (int q = 0; q < H / 4; ++q) {
        float4 a = wh[q], hh = hv[q];
        acc += a.x * hh.x + a.y * hh.y + a.z * hh.z + a.w * hh.w;
      }
      gates[tid] = acc;
    }
    __syncthreads();
    if (tid < H) {
      float ig = sigf(gates[tid]);
      float fg = sigf(gates[H + tid]);
      float gg = tanhf(gates[2 * H + tid]);
      float og = sigf(gates[3 * H + tid]);
      float cn = fg * c[tid] + ig * gg;
      float hn = og * tanhf(cn);
      c[tid] = cn; h[tid] = hn;
      h16[(size_t)t * H + tid] = (_Float16)hn;   // f16 A-matrix for WMMA GEMM
    }
    __syncthreads();
  }
}

// ---------------- Wout f32 -> f16, zero-padded to VP rows ----------------
__global__ __launch_bounds__(512) void cvt_kernel(
    const float* __restrict__ Wout, _Float16* __restrict__ w16)
{
  const int idx = blockIdx.x * 512 + threadIdx.x;        // covers VP*H exactly
  w16[idx] = (idx < V * H) ? (_Float16)Wout[idx] : (_Float16)0.0f;
}

// ---------------- rowBest init ----------------
__global__ __launch_bounds__(512) void init_kernel(unsigned long long* __restrict__ rowBest)
{
  const int i = blockIdx.x * 512 + threadIdx.x;
  if (i < T) rowBest[i] = 0ull;
}

// ---------------- preds = Hdec @ Wout^T + bout via WMMA f16, fused argmax ---
// One wave = one 16x16 D tile. 8 waves/block share the same N-tile (B reuse).
// Stores are non-temporal (205 MB stream) so f16 Wout stays L2-resident.
// Per-row argmax folded in: shuffle-reduce 16 columns -> 1 atomic_max_u64/row.
__global__ __launch_bounds__(256) void preds_kernel(
    const _Float16* __restrict__ h16, const _Float16* __restrict__ w16,
    const float* __restrict__ bout, float* __restrict__ out,
    unsigned long long* __restrict__ rowBest)
{
  const int wave  = threadIdx.x >> 5;
  const int lane  = threadIdx.x & 31;
  const int tileN = blockIdx.x;
  const int tileM = blockIdx.y * 8 + wave;

  const int mrow = tileM * 16 + (lane & 15);             // A row (M) for this lane
  const int ncol = tileN * 16 + (lane & 15);             // B col (N) for this lane
  const int lo   = (lane & 16) ? 16 : 0;                 // A K-offset (contiguous, matches B)
  const int bo   = (lane & 16) ? 16 : 0;                 // B K-offset (contiguous, matches A)

  const _Float16* Ap = h16 + (size_t)mrow * H;
  const _Float16* Bp = w16 + (size_t)ncol * H;

  v8f acc = {};
  #pragma unroll
  for (int kb = 0; kb < H; kb += 32) {
    v8h a0 = *(const v8h*)(Ap + kb + lo);                // K = kb+lo .. +7
    v8h a1 = *(const v8h*)(Ap + kb + 8 + lo);            // K = kb+8+lo .. +7
    v16h a = __builtin_shufflevector(a0, a1, 0,1,2,3,4,5,6,7,8,9,10,11,12,13,14,15);
    v16h bv = *(const v16h*)(Bp + kb + bo);              // K = kb+bo .. +15 (contig)
    acc = __builtin_amdgcn_wmma_f32_16x16x32_f16(
        false, a, false, bv, (short)0, acc, false, false);
  }

  const bool valid = (ncol < V);                         // only last N-tile diverges
  const float bias = valid ? bout[ncol] : 0.0f;
  const int mbase = tileM * 16 + ((lane & 16) ? 8 : 0);  // D row base for this half

  #pragma unroll
  for (int r = 0; r < 8; ++r) {
    const float val = acc[r] + bias;
    if (valid)
      __builtin_nontemporal_store(val, &out[(size_t)(mbase + r) * V + ncol]);

    // pack (orderable value, first-index tie-break) into a 64-bit max key
    unsigned long long key = 0ull;
    if (valid) {
      unsigned u   = __float_as_uint(val);
      unsigned ord = (u & 0x80000000u) ? ~u : (u | 0x80000000u);
      key = ((unsigned long long)ord << 32) |
            (unsigned long long)(0xFFFFFFFFu - (unsigned)ncol);
    }
    // reduce over the 16 columns held by this 16-lane half (all lanes active)
    #pragma unroll
    for (int m = 1; m < 16; m <<= 1) {
      unsigned long long o = __shfl_xor(key, m, 32);
      if (o > key) key = o;
    }
    if ((lane & 15) == 0)
      atomicMax(&rowBest[mbase + r], key);
  }
}

// ---------------- decode argmax keys -> float indices ----------------
__global__ __launch_bounds__(512) void decode_kernel(
    const unsigned long long* __restrict__ rowBest, float* __restrict__ outIdx)
{
  const int i = blockIdx.x * 512 + threadIdx.x;
  if (i < T) {
    unsigned low = (unsigned)(rowBest[i] & 0xFFFFFFFFull);
    outIdx[i] = (float)(0xFFFFFFFFu - low);
  }
}

// ---------------- launch ----------------
extern "C" void kernel_launch(void* const* d_in, const int* in_sizes, int n_in,
                              void* d_out, int out_size, void* d_ws, size_t ws_size,
                              hipStream_t stream) {
  const int*   input_seq = (const int*)  d_in[0];
  const int*   gold_seq  = (const int*)  d_in[1];
  const float* embeds    = (const float*)d_in[2];
  const float* Wih_enc   = (const float*)d_in[3];
  const float* Whh_enc   = (const float*)d_in[4];
  const float* b_enc     = (const float*)d_in[5];
  const float* Wih_dec   = (const float*)d_in[6];
  const float* Whh_dec   = (const float*)d_in[7];
  const float* b_dec     = (const float*)d_in[8];
  const float* W1        = (const float*)d_in[9];
  const float* W2        = (const float*)d_in[10];
  const float* v         = (const float*)d_in[11];
  const float* Wout      = (const float*)d_in[12];
  const float* bout      = (const float*)d_in[13];

  float* out = (float*)d_out;                      // [T*V] preds, then [T] argmax

  // workspace layout (all 16B aligned)
  float*    enc_out = (float*)d_ws;                           // T*H f32
  float*    encW1   = enc_out + (size_t)T * H;                // T*H f32
  float*    h0c0    = encW1   + (size_t)T * H;                // 2*H f32
  _Float16* h16     = (_Float16*)(h0c0 + 2 * H);              // T*H f16
  _Float16* w16     = h16 + (size_t)T * H;                    // VP*H f16 (~12.9 MB)
  unsigned long long* rowBest =
      (unsigned long long*)(w16 + (size_t)VP * H);            // T u64 (8 KB)

  // independent prep: Wout f32->f16 (padded) + argmax slots
  cvt_kernel<<<dim3((VP * H) / 512), 512, 0, stream>>>(Wout, w16);
  init_kernel<<<dim3((T + 511) / 512), 512, 0, stream>>>(rowBest);
  // sequential phases
  enc_kernel<<<1, 512, 0, stream>>>(input_seq, embeds, Wih_enc, Whh_enc, b_enc,
                                    enc_out, h0c0);
  encw1_kernel<<<dim3(T), H, 0, stream>>>(enc_out, W1, encW1);
  dec_kernel<<<1, 512, 0, stream>>>(gold_seq, embeds, Wih_dec, Whh_dec, b_dec,
                                    W2, v, enc_out, encW1, h0c0, h16);
  // dominant GEMM (WMMA) with fused argmax, then decode indices
  preds_kernel<<<dim3(NT, MT / 8), 256, 0, stream>>>(h16, w16, bout, out, rowBest);
  decode_kernel<<<dim3((T + 511) / 512), 512, 0, stream>>>(rowBest, out + (size_t)T * V);
}
